// WindowsInt4Linear_35347580846659
// MI455X (gfx1250) — matrix operand order repaired and checked by reading, not compiled, not run on hardware
//
#include <hip/hip_runtime.h>
#include <hip/hip_bf16.h>

// ---------------------------------------------------------------------------
// Int4 group-quantized linear: out[4096,11008] = X[4096,4096] * W^T
//   W[n,k] = ((packed>> (k even?4:0)) & 0xF) * scale[n,k/128] + zero[n,k/128]
// Compute-bound (AI ~1085 FLOP/B >> machine balance) -> f16 WMMA, f32 accum.
// Magic-number dequant: as_f16(0x6400|q) = 1024+q exactly, so
//   w = f16(q) * s + (z - 1024*s)   via one v_pk_fma_f16 per weight pair.
// ---------------------------------------------------------------------------

typedef __attribute__((ext_vector_type(16))) _Float16 v16h;
typedef __attribute__((ext_vector_type(8)))  _Float16 v8h;
typedef __attribute__((ext_vector_type(4)))  _Float16 v4h;
typedef __attribute__((ext_vector_type(2)))  _Float16 h2;
typedef __attribute__((ext_vector_type(8)))  float    v8f;
typedef __attribute__((ext_vector_type(4)))  float    f4;
typedef __attribute__((ext_vector_type(4)))  int      i4;

#define IN_F   4096
#define OUT_F  11008
#define GS     128
#define M_TOT  4096
#define BM     128
#define BN     128
#define BK     64
#define LDSW   72          // padded row stride in halves (144 B) -> bank-friendly
#define NKT    (IN_F / BK) // 64 k-tiles

__global__ __launch_bounds__(256) void int4_linear_wmma_kernel(
    const float* __restrict__ x,        // [4096, 4096] f32
    const int*   __restrict__ pw,       // [11008, 2048] one byte per int32
    const float* __restrict__ scale,    // [11008, 32]
    const float* __restrict__ zp,       // [11008, 32]
    float*       __restrict__ out)      // [4096, 11008]
{
  __shared__ _Float16 Xs[BM * LDSW];    // 18 KB
  __shared__ _Float16 Ws[BN * LDSW];    // 18 KB

  const int tid  = threadIdx.x;
  const int m0   = blockIdx.x * BM;
  const int n0   = blockIdx.y * BN;

  const int wave = tid >> 5;
  const int lane = tid & 31;
  const int wm   = wave & 3;            // 4 waves along M (32 rows each)
  const int wn   = wave >> 2;           // 2 waves along N (64 cols each)
  const int lr   = lane & 15;
  const int lh   = lane >> 4;           // half-wave select

  v8f acc[2][4];
#pragma unroll
  for (int i = 0; i < 2; ++i)
#pragma unroll
    for (int j = 0; j < 4; ++j) acc[i][j] = (v8f)0.f;

  // ---- register staging for the software pipeline ----
  f4    xreg[8];                        // 128x64 f32 X tile: 8 float4 / thread
  i4    wreg[4];                        // 128x32 packed bytes: 4 int4 / thread
  float sreg[4], zreg[4];

  auto load_tile = [&](int kt) {
    const int k0 = kt * BK;
#pragma unroll
    for (int i = 0; i < 8; ++i) {
      const int c = tid + 256 * i;      // fully coalesced float4 chunks
      const int row = c >> 4, col = c & 15;
      xreg[i] = *(const f4*)(x + (size_t)(m0 + row) * IN_F + k0 + col * 4);
    }
    const int g = k0 / GS;              // whole k-tile lies in one quant group
#pragma unroll
    for (int i = 0; i < 4; ++i) {
      const int c = tid + 256 * i;      // fully coalesced int4 chunks
      const int row = c >> 3, cj = c & 7;
      wreg[i] = *(const i4*)(pw + (size_t)(n0 + row) * (IN_F / 2) + (k0 >> 1) + cj * 4);
      sreg[i] = scale[(n0 + row) * (IN_F / GS) + g];
      zreg[i] = zp[(n0 + row) * (IN_F / GS) + g];
    }
  };

  auto store_tile = [&]() {
    // X: f32 -> f16, 4 halves (8 B) per store
#pragma unroll
    for (int i = 0; i < 8; ++i) {
      const int c = tid + 256 * i;
      const int row = c >> 4, col = c & 15;
      v4h v = { (_Float16)xreg[i].x, (_Float16)xreg[i].y,
                (_Float16)xreg[i].z, (_Float16)xreg[i].w };
      *(v4h*)(&Xs[row * LDSW + col * 4]) = v;
    }
    // W: magic-number dequant, 8 halves (16 B) per store
#pragma unroll
    for (int i = 0; i < 4; ++i) {
      const int c = tid + 256 * i;
      const int row = c >> 3, cj = c & 7;
      const float    s  = sreg[i];
      const _Float16 sh = (_Float16)s;
      const _Float16 zh = (_Float16)(zreg[i] - 1024.0f * s);
      const h2 s2 = { sh, sh };
      const h2 z2 = { zh, zh };
      h2 w[4];
#pragma unroll
      for (int j = 0; j < 4; ++j) {
        const unsigned u = (unsigned)wreg[i][j];   // one byte: hi nibble = k even
        const unsigned bits = 0x64006400u | ((u >> 4) & 0xFu) | ((u & 0xFu) << 16);
        const h2 hv = __builtin_bit_cast(h2, bits);
        w[j] = hv * s2 + z2;                       // v_pk_fma_f16
      }
      v8h ov = { w[0].x, w[0].y, w[1].x, w[1].y,
                 w[2].x, w[2].y, w[3].x, w[3].y };
      *(v8h*)(&Ws[row * LDSW + cj * 8]) = ov;
    }
  };

  auto compute = [&]() {
#pragma unroll
    for (int ks = 0; ks < 2; ++ks) {               // two K=32 steps per tile
      const int kb = ks * 32;
      v16h a[2], b[4];
      // A fragment: lanes<16 -> K{0-7,16-23}, lanes>=16 -> K{8-15,24-31}
#pragma unroll
      for (int mi = 0; mi < 2; ++mi) {
        const int row = wm * 32 + mi * 16 + lr;
        const _Float16* base = &Xs[row * LDSW + kb + lh * 8];
        v8h p0 = *(const v8h*)(base);
        v8h p1 = *(const v8h*)(base + 16);
        a[mi] = __builtin_shufflevector(p0, p1, 0,1,2,3,4,5,6,7,8,9,10,11,12,13,14,15);
      }
      // B fragment: lanes<16 -> K{0-7,16-23}, lanes>=16 -> K{8-15,24-31}
#pragma unroll
      for (int ni = 0; ni < 4; ++ni) {
        const int col = wn * 64 + ni * 16 + lr;
        const _Float16* base = &Ws[col * LDSW + kb + lh * 8];
        v8h p0 = *(const v8h*)(base);
        v8h p1 = *(const v8h*)(base + 16);
        b[ni] = __builtin_shufflevector(p0, p1, 0,1,2,3,4,5,6,7,8,9,10,11,12,13,14,15);
      }
#pragma unroll
      for (int mi = 0; mi < 2; ++mi)
#pragma unroll
        for (int ni = 0; ni < 4; ++ni)
          acc[mi][ni] = __builtin_amdgcn_wmma_f32_16x16x32_f16(
              false, a[mi], false, b[ni], (short)0, acc[mi][ni], false, false);
    }
  };

  load_tile(0);
  for (int kt = 0; kt < NKT; ++kt) {
    __syncthreads();          // all waves done reading LDS from previous iter
    store_tile();
    __syncthreads();          // LDS tile visible to all waves
    if (kt + 1 < NKT) load_tile(kt + 1);  // overlap global latency with WMMA
    compute();
  }

  // C/D layout: lane<16 -> N=lr,M=v ; lane>=16 -> N=lr,M=8+v
#pragma unroll
  for (int mi = 0; mi < 2; ++mi)
#pragma unroll
    for (int ni = 0; ni < 4; ++ni) {
      const int n     = n0 + wn * 64 + ni * 16 + lr;
      const int mbase = m0 + wm * 32 + mi * 16 + lh * 8;
#pragma unroll
      for (int v = 0; v < 8; ++v)
        out[(size_t)(mbase + v) * OUT_F + n] = acc[mi][ni][v];
    }
}

extern "C" void kernel_launch(void* const* d_in, const int* in_sizes, int n_in,
                              void* d_out, int out_size, void* d_ws, size_t ws_size,
                              hipStream_t stream) {
  const float* x  = (const float*)d_in[0];
  const int*   pw = (const int*)d_in[1];
  const float* sc = (const float*)d_in[2];
  const float* zp = (const float*)d_in[3];
  float* out = (float*)d_out;

  dim3 grid(M_TOT / BM, OUT_F / BN);   // (32, 86); x-major order groups blocks
  int4_linear_wmma_kernel<<<grid, 256, 0, stream>>>(x, pw, sc, zp, out);
}
